// SRD_51754355916944
// MI455X (gfx1250) — matrix operand order repaired and checked by reading, not compiled, not run on hardware
//
#include <hip/hip_runtime.h>
#include <hip/hip_bf16.h>
#include <math.h>

typedef __attribute__((ext_vector_type(16))) _Float16 v16h;
typedef __attribute__((ext_vector_type(8)))  float    v8f;

#define BB   16
#define HH   512
#define WW   512
#define HWPIX (HH*WW)          // 262144
#define NPIX  (BB*HWPIX)       // 4194304
#define NBINS 4096             // 16^3

// Gaussian taps (precomputed, normalized)
__constant__ float G5[5]  = {0.0544886845f, 0.2442013420f, 0.4026199469f,
                             0.2442013420f, 0.0544886845f};
__constant__ float G11[11] = {0.0010284844f, 0.0075988158f, 0.0360007770f,
                              0.1093606027f, 0.2130055295f, 0.2660115811f,
                              0.2130055295f, 0.1093606027f, 0.0360007770f,
                              0.0075988158f, 0.0010284844f};
__constant__ int OFFY[8] = {0, 1, 1,  1,  0, -1, -1, -1};
__constant__ int OFFX[8] = {1, 1, 0, -1, -1, -1,  0,  1};

__device__ __forceinline__ int reflect_i(int i, int n) {
  i = (i < 0) ? -i : i;
  i = (i >= n) ? (2*n - 2 - i) : i;
  return i;
}
__device__ __forceinline__ int clamp_i(int i, int n) {
  return i < 0 ? 0 : (i >= n ? n - 1 : i);
}
// order-preserving float<->uint for atomic min/max
__device__ __forceinline__ unsigned int f2o(float f) {
  unsigned int b = __float_as_uint(f);
  return (b & 0x80000000u) ? ~b : (b | 0x80000000u);
}
__device__ __forceinline__ float o2f(unsigned int u) {
  return (u & 0x80000000u) ? __uint_as_float(u & 0x7FFFFFFFu)
                           : __uint_as_float(~u);
}
// async Global->LDS copy, one b32 per active lane (ASYNCcnt tracked)
__device__ __forceinline__ void async_g2l_b32(unsigned int lds_addr, const void* gptr) {
  unsigned long long ga = (unsigned long long)(size_t)gptr;
  asm volatile("global_load_async_to_lds_b32 %0, %1, off"
               :: "v"(lds_addr), "v"(ga) : "memory");
}
__device__ __forceinline__ void wait_async0() {
  asm volatile("s_wait_asynccnt 0x0" ::: "memory");
}

// ---------------- init: zero histogram, seed min/max ----------------
__global__ void k_init(unsigned int* hist, unsigned int* mm) {
  int t = blockIdx.x * 256 + threadIdx.x;
  if (t < BB * NBINS) hist[t] = 0u;
  else if (t < BB * NBINS + 64) {
    int j = t - BB * NBINS;   // [0,16) edmin [16,32) edmax [32,48) spmin [48,64) spmax
    mm[j] = (j < 16 || (j >= 32 && j < 48)) ? 0xFFFFFFFFu : 0u;
  }
}

// -------- denorm+clip, gray, quantized color index, LDS-aggregated histogram --------
__global__ void k_gray_hist(const float* __restrict__ xn,
                            const float* __restrict__ mean,
                            const float* __restrict__ stdv,
                            float* __restrict__ gray,
                            int* __restrict__ idx,
                            unsigned int* __restrict__ hist) {
  __shared__ unsigned int lh[NBINS];
  int tid = threadIdx.x;
  for (int i = tid; i < NBINS; i += 256) lh[i] = 0u;
  __syncthreads();
  int p = blockIdx.x * 256 + tid;               // [0, NPIX); block covers one image only
  int b = p / HWPIX, pos = p - b * HWPIX;
  int base = b * 3 * HWPIX + pos;
  __builtin_prefetch(xn + base + 2048, 0, 1);
  float x0 = fminf(fmaxf(xn[base          ] * stdv[0] + mean[0], 0.f), 1.f);
  float x1 = fminf(fmaxf(xn[base +   HWPIX] * stdv[1] + mean[1], 0.f), 1.f);
  float x2 = fminf(fmaxf(xn[base + 2*HWPIX] * stdv[2] + mean[2], 0.f), 1.f);
  gray[p] = 0.299f * x0 + 0.587f * x1 + 0.114f * x2;
  int q0 = (int)rintf(x0 * 15.f);
  int q1 = (int)rintf(x1 * 15.f);
  int q2 = (int)rintf(x2 * 15.f);
  int c = q0 * 256 + q1 * 16 + q2;
  idx[p] = c;
  atomicAdd(&lh[c], 1u);                        // ds_add, banked
  __syncthreads();
  for (int i = tid; i < NBINS; i += 256) {
    unsigned int v = lh[i];
    if (v) atomicAdd(&hist[b * NBINS + i], v);  // only non-zero bins hit HBM/L2
  }
}

// ---------------- 5-tap separable Gaussian (reflect) ----------------
__global__ void k_blur5_h(const float* __restrict__ in, float* __restrict__ out) {
  int p = blockIdx.x * 256 + threadIdx.x;
  int b = p / HWPIX, pos = p - b * HWPIX;
  int y = pos / WW, x = pos - y * WW;
  const float* img = in + b * HWPIX + y * WW;
  float s = 0.f;
  #pragma unroll
  for (int t = 0; t < 5; ++t) s += G5[t] * img[reflect_i(x + t - 2, WW)];
  out[p] = s;
}
__global__ void k_blur5_v(const float* __restrict__ in, float* __restrict__ out) {
  int p = blockIdx.x * 256 + threadIdx.x;
  int b = p / HWPIX, pos = p - b * HWPIX;
  int y = pos / WW, x = pos - y * WW;
  const float* img = in + b * HWPIX;
  float s = 0.f;
  #pragma unroll
  for (int t = 0; t < 5; ++t) s += G5[t] * img[reflect_i(y + t - 2, HH) * WW + x];
  out[p] = s;
}

// ---------------- Sobel (edge/clamp pad), mag + direction ----------------
__global__ void k_sobel(const float* __restrict__ blur,
                        float* __restrict__ mag, unsigned char* __restrict__ dir) {
  int p = blockIdx.x * 256 + threadIdx.x;
  int b = p / HWPIX, pos = p - b * HWPIX;
  int y = pos / WW, x = pos - y * WW;
  const float* img = blur + b * HWPIX;
  int ym = clamp_i(y-1, HH), yp = clamp_i(y+1, HH);
  int xm = clamp_i(x-1, WW), xp = clamp_i(x+1, WW);
  float a00 = img[ym*WW+xm], a01 = img[ym*WW+x], a02 = img[ym*WW+xp];
  float a10 = img[y *WW+xm],                    a12 = img[y *WW+xp];
  float a20 = img[yp*WW+xm], a21 = img[yp*WW+x], a22 = img[yp*WW+xp];
  float gx = (a02 - a00) + 2.f*(a12 - a10) + (a22 - a20);
  float gy = (a20 - a00) + 2.f*(a21 - a01) + (a22 - a02);
  mag[p] = sqrtf(gx*gx + gy*gy + 1e-6f);
  float deg = atan2f(gy, gx) * 57.29577951308232f;
  int ang = (int)rintf(deg * (1.f/45.f));       // in [-4,4]
  dir[p] = (unsigned char)((ang + 8) & 7);
}

// ---------------- NMS + double threshold -> state {0,1,2} ----------------
__global__ void k_nms(const float* __restrict__ mag,
                      const unsigned char* __restrict__ dir,
                      unsigned char* __restrict__ st) {
  int p = blockIdx.x * 256 + threadIdx.x;
  int b = p / HWPIX, pos = p - b * HWPIX;
  int y = pos / WW, x = pos - y * WW;
  const float* m = mag + b * HWPIX;
  float mc = m[pos];
  int d  = dir[p];
  int d2 = (d + 4) & 7;
  int y1 = y + OFFY[d],  x1 = x + OFFX[d];
  int y2 = y + OFFY[d2], x2 = x + OFFX[d2];
  float n1 = (y1 >= 0 && y1 < HH && x1 >= 0 && x1 < WW) ? m[y1*WW + x1] : 0.f;
  float n2 = (y2 >= 0 && y2 < HH && x2 >= 0 && x2 < WW) ? m[y2*WW + x2] : 0.f;
  float kept = (fminf(mc - n1, mc - n2) > 0.f) ? mc : 0.f;
  st[p] = kept > 0.2f ? (unsigned char)2 : (kept > 0.1f ? (unsigned char)1 : (unsigned char)0);
}

// ------- hysteresis: tile-local flood fill in LDS, monotone in-place -------
__global__ void k_hyst(unsigned char* __restrict__ st) {
  __shared__ unsigned char s[34][34];
  __shared__ int changed;
  int tx = threadIdx.x, ty = threadIdx.y;             // 32 x 8
  int tid = ty * 32 + tx;
  int b = blockIdx.z;
  int gx0 = blockIdx.x * 32 - 1, gy0 = blockIdx.y * 32 - 1;
  unsigned char* base = st + (size_t)b * HWPIX;
  for (int i = tid; i < 34*34; i += 256) {
    int ly = i / 34, lx = i - ly * 34;
    int gy = gy0 + ly, gx = gx0 + lx;
    s[ly][lx] = (gy >= 0 && gy < HH && gx >= 0 && gx < WW) ? base[gy*WW + gx] : (unsigned char)0;
  }
  __syncthreads();
  for (int it = 0; it < 64; ++it) {
    if (tid == 0) changed = 0;
    __syncthreads();
    int local = 0;
    #pragma unroll
    for (int ry = 0; ry < 4; ++ry) {
      int ly = ty * 4 + ry + 1, lx = tx + 1;
      if (s[ly][lx] == 1) {
        int strong = (s[ly-1][lx-1]==2) | (s[ly-1][lx]==2) | (s[ly-1][lx+1]==2) |
                     (s[ly  ][lx-1]==2) |                    (s[ly  ][lx+1]==2) |
                     (s[ly+1][lx-1]==2) | (s[ly+1][lx]==2) | (s[ly+1][lx+1]==2);
        if (strong) { s[ly][lx] = 2; local = 1; }
      }
    }
    if (local) changed = 1;
    __syncthreads();
    int done = (changed == 0);
    __syncthreads();
    if (done) break;
  }
  #pragma unroll
  for (int ry = 0; ry < 4; ++ry) {
    int ly = ty * 4 + ry + 1;
    int gy = gy0 + ly, gx = gx0 + tx + 1;             // interior: always in range
    base[gy*WW + gx] = s[ly][tx + 1];
  }
}

__global__ void k_final_edges(const unsigned char* __restrict__ st, float* __restrict__ e) {
  int p = blockIdx.x * 256 + threadIdx.x;
  e[p] = (st[p] == 2) ? 1.f : 0.f;
}

// ------- 11-tap separable blur on matrix cores: D(16x16)=A(16x32)xB(32x16) -------
// Patch staged Global->LDS with async copy engine; per-wave private LDS slot.
// Horizontal: A = image patch (cols c0-8..c0+23), B[k][n] = g11[k-n-3] band.
__global__ void k_density_h(const float* __restrict__ in, float* __restrict__ out) {
  __shared__ float sb[8][512];                        // 8 waves x (16 rows x 32 cols) f32
  int w = threadIdx.x >> 5, lane = threadIdx.x & 31;
  int gw = blockIdx.x * 8 + w;                        // tile id
  int b  = gw >> 10;                                  // 1024 tiles/image
  int ti = gw & 1023;
  int r0 = (ti >> 5) << 4, c0 = (ti & 31) << 4;
  const float* img = in + (size_t)b * HWPIX;
  unsigned int lds0 = (unsigned int)(size_t)(&sb[w][0]);  // LDS byte addr = low 32 bits
  #pragma unroll
  for (int it = 0; it < 16; ++it) {
    int e   = it * 32 + lane;                         // element in patch
    int row = e >> 5, col = e & 31;
    async_g2l_b32(lds0 + (unsigned int)(e * 4),
                  img + (size_t)(r0 + row) * WW + reflect_i(c0 - 8 + col, WW));
  }
  wait_async0();
  int half = lane >> 4, m = lane & 15;
  v16h a, bm;
  #pragma unroll
  for (int j = 0; j < 16; ++j) {
    int k = (j < 8 ? j : j + 8) + half * 8;           // A K-layout (16-bit, 16x32)
    a[j] = (_Float16)sb[w][m * 32 + k];
    int kb = (j < 8 ? j : j + 8) + half * 8;          // B K-layout (16-bit, 32x16)
    int t = kb - m - 3;                               // n = lane&15
    bm[j] = (_Float16)((t >= 0 && t < 11) ? G11[t] : 0.f);
  }
  v8f c = {};
  c = __builtin_amdgcn_wmma_f32_16x16x32_f16(false, a, false, bm, (short)0, c, false, false);
  float* o = out + (size_t)b * HWPIX;
  #pragma unroll
  for (int v = 0; v < 8; ++v)
    o[(r0 + v + half * 8) * WW + (c0 + m)] = c[v];    // D: M = v + 8*half, N = lane&15
}
// Vertical: A[m][k] = g11[k-m-3] band, B = image patch (rows r0-8..r0+23).
__global__ void k_density_v(const float* __restrict__ in, float* __restrict__ out) {
  __shared__ float sb[8][512];                        // 8 waves x (32 rows x 16 cols) f32
  int w = threadIdx.x >> 5, lane = threadIdx.x & 31;
  int gw = blockIdx.x * 8 + w;
  int b  = gw >> 10;
  int ti = gw & 1023;
  int r0 = (ti >> 5) << 4, c0 = (ti & 31) << 4;
  const float* img = in + (size_t)b * HWPIX;
  unsigned int lds0 = (unsigned int)(size_t)(&sb[w][0]);
  #pragma unroll
  for (int it = 0; it < 16; ++it) {
    int e   = it * 32 + lane;
    int row = e >> 4, col = e & 15;
    async_g2l_b32(lds0 + (unsigned int)(e * 4),
                  img + (size_t)reflect_i(r0 - 8 + row, HH) * WW + (c0 + col));
  }
  wait_async0();
  int half = lane >> 4, m = lane & 15;
  v16h a, bm;
  #pragma unroll
  for (int j = 0; j < 16; ++j) {
    int k = half * 16 + j;
    int t = k - m - 3;
    a[j] = (_Float16)((t >= 0 && t < 11) ? G11[t] : 0.f);
    int kb = half * 16 + j;
    bm[j] = (_Float16)sb[w][kb * 16 + m];
  }
  v8f c = {};
  c = __builtin_amdgcn_wmma_f32_16x16x32_f16(false, a, false, bm, (short)0, c, false, false);
  float* o = out + (size_t)b * HWPIX;
  #pragma unroll
  for (int v = 0; v < 8; ++v)
    o[(r0 + v + half * 8) * WW + (c0 + m)] = c[v];
}

// ---------------- sparsity map ----------------
__global__ void k_sparsity(const int* __restrict__ idx,
                           const unsigned int* __restrict__ hist,
                           float* __restrict__ sp) {
  int p = blockIdx.x * 256 + threadIdx.x;
  int b = p / HWPIX;
  float prob = (float)hist[b * NBINS + idx[p]] * (1.f / (float)HWPIX);
  sp[p] = -logf(prob + 1e-9f);
}

// ---------------- per-image min/max reduction ----------------
__global__ void k_minmax(const float* __restrict__ m,
                         unsigned int* __restrict__ mn, unsigned int* __restrict__ mx) {
  __shared__ float smn[256], smx[256];
  int b = blockIdx.y, tid = threadIdx.x;
  const float* p = m + (size_t)b * HWPIX;
  float lmin = 3.4e38f, lmax = -3.4e38f;
  for (int i = blockIdx.x * 256 + tid; i < HWPIX; i += gridDim.x * 256) {
    float v = p[i];
    lmin = fminf(lmin, v); lmax = fmaxf(lmax, v);
  }
  smn[tid] = lmin; smx[tid] = lmax;
  __syncthreads();
  for (int s = 128; s > 0; s >>= 1) {
    if (tid < s) { smn[tid] = fminf(smn[tid], smn[tid+s]); smx[tid] = fmaxf(smx[tid], smx[tid+s]); }
    __syncthreads();
  }
  if (tid == 0) { atomicMin(&mn[b], f2o(smn[0])); atomicMax(&mx[b], f2o(smx[0])); }
}

// ---------------- normalize, fuse, sigmoid ----------------
__global__ void k_fused(const float* __restrict__ ed, const float* __restrict__ sp,
                        const unsigned int* __restrict__ mm,
                        const float* __restrict__ alpha, const float* __restrict__ beta,
                        float* __restrict__ out) {
  int p = blockIdx.x * 256 + threadIdx.x;
  int b = p / HWPIX;
  float edmin = o2f(mm[b]),      edmax = o2f(mm[16 + b]);
  float spmin = o2f(mm[32 + b]), spmax = o2f(mm[48 + b]);
  float ned = (ed[p] - edmin) / (edmax - edmin + 1e-9f);
  float nsp = (sp[p] - spmin) / (spmax - spmin + 1e-9f);
  float f = alpha[0] * ned + beta[0] * nsp;
  out[p] = 1.f / (1.f + expf(-f));
}

extern "C" void kernel_launch(void* const* d_in, const int* in_sizes, int n_in,
                              void* d_out, int out_size, void* d_ws, size_t ws_size,
                              hipStream_t stream) {
  const float* xn    = (const float*)d_in[0];
  const float* mean  = (const float*)d_in[1];
  const float* stdv  = (const float*)d_in[2];
  const float* alpha = (const float*)d_in[3];
  const float* beta  = (const float*)d_in[4];
  float* out = (float*)d_out;

  // workspace layout
  float* grayF = (float*)d_ws;                 // NPIX
  float* tmpF  = grayF + NPIX;                 // NPIX
  float* blurF = tmpF  + NPIX;                 // NPIX (later reused as edgesF)
  float* magF  = blurF + NPIX;                 // NPIX
  float* edF   = magF  + NPIX;                 // NPIX
  float* spF   = edF   + NPIX;                 // NPIX
  int*   idxI  = (int*)(spF + NPIX);           // NPIX
  unsigned char* dirU = (unsigned char*)(idxI + NPIX);   // NPIX bytes
  unsigned char* stU  = dirU + NPIX;                     // NPIX bytes
  unsigned int* hist  = (unsigned int*)(stU + NPIX);     // BB*NBINS
  unsigned int* mm    = hist + BB * NBINS;               // 64

  const int EB = NPIX / 256;                   // element-wise grid (16384)

  k_init<<<(BB*NBINS + 64 + 255)/256, 256, 0, stream>>>(hist, mm);
  k_gray_hist<<<EB, 256, 0, stream>>>(xn, mean, stdv, grayF, idxI, hist);
  k_blur5_h<<<EB, 256, 0, stream>>>(grayF, tmpF);
  k_blur5_v<<<EB, 256, 0, stream>>>(tmpF, blurF);
  k_sobel<<<EB, 256, 0, stream>>>(blurF, magF, dirU);
  k_nms<<<EB, 256, 0, stream>>>(magF, dirU, stU);
  for (int i = 0; i < 64; ++i)
    k_hyst<<<dim3(WW/32, HH/32, BB), dim3(32, 8), 0, stream>>>(stU);
  k_final_edges<<<EB, 256, 0, stream>>>(stU, blurF);     // blurF := edges as float
  // WMMA tensor-core 11x11 separable blur: 1 wave per 16x16 tile, async LDS staging
  k_density_h<<<(BB*1024*32)/256, 256, 0, stream>>>(blurF, tmpF);
  k_density_v<<<(BB*1024*32)/256, 256, 0, stream>>>(tmpF, edF);
  k_sparsity<<<EB, 256, 0, stream>>>(idxI, hist, spF);
  k_minmax<<<dim3(32, BB), 256, 0, stream>>>(edF, mm +  0, mm + 16);
  k_minmax<<<dim3(32, BB), 256, 0, stream>>>(spF, mm + 32, mm + 48);
  k_fused<<<EB, 256, 0, stream>>>(edF, spF, mm, alpha, beta, out);
}
